// layers_40999757808045
// MI455X (gfx1250) — compile-verified, matrix operand order checked by reading, not run on hardware
//
#include <hip/hip_runtime.h>
#include <hip/hip_bf16.h>

// GAT layer, algebraically reduced:
//   out[h,i,:] = (adj @ (e2_h * mx_h)) / (adj @ e2_h),  e2 = exp(mx·alpha)
// (exp(c1_i) cancels in the softmax; alpha_origin is mathematically unused.)
// Dominant HBM cost: streaming adj (64 MB f32) once -> ~2.9us at 23.3 TB/s.
// Bt (4.5 MB, L2-resident) is staged per-block into LDS via async
// global->LDS copies (double-buffered, ASYNCcnt-tracked) so the 8 waves of a
// block share fragments instead of 8x-re-reading L2.
// All GEMMs run on v_wmma_f32_16x16x32_bf16 with hi/lo bf16 splitting for
// near-f32 accuracy (adj is 0/1 -> exact in bf16).

#define NN   4096
#define FIN  512
#define NH   4
#define DH   64
#define OUTC 256   // NH*DH
#define CPAD 288   // 256 feature cols + 4 denom cols + 28 zero pad (18 tiles)
#define NT   18    // CPAD/16
#define LCS  40    // LDS col stride in ushorts (pad 32->40: conflict-free frag reads)

typedef __attribute__((ext_vector_type(16))) __bf16 v16bf;
typedef __attribute__((ext_vector_type(8)))  float  v8f;
typedef __attribute__((ext_vector_type(4)))  int    v4i;

union BFrag { v16bf v; uint4 q[2]; unsigned int u[8]; };
union Acc   { v8f v; float f[8]; };

__device__ __forceinline__ unsigned short f2bf(float f) {
    unsigned int u = __float_as_uint(f);
    u += 0x7FFFu + ((u >> 16) & 1u);           // round-to-nearest-even
    return (unsigned short)(u >> 16);
}
__device__ __forceinline__ float bf2f(unsigned short s) {
    return __uint_as_float(((unsigned int)s) << 16);
}
// truncating pack: exact for adj's 0.0/1.0 values; single v_perm-able op
__device__ __forceinline__ unsigned int pack2_trunc(float a, float b) {
    return (__float_as_uint(a) >> 16) | (__float_as_uint(b) & 0xFFFF0000u);
}

// ---- async global -> LDS copy (CDNA5), with portable fallback -----------
#if __has_builtin(__builtin_amdgcn_global_load_async_to_lds_b128)
#define HAVE_ASYNC_LDS 1
__device__ __forceinline__ void async_cp16(const void* g, void* l) {
    __builtin_amdgcn_global_load_async_to_lds_b128(
        (__attribute__((address_space(1))) v4i*)(void*)g,
        (__attribute__((address_space(3))) v4i*)l, 0, 0);
}
#else
#define HAVE_ASYNC_LDS 0
__device__ __forceinline__ void async_cp16(const void* g, void* l) {
    *(uint4*)l = *(const uint4*)g;             // global_load + ds_store
}
#endif

template <int N>
__device__ __forceinline__ void async_wait() {
#if HAVE_ASYNC_LDS
#if __has_builtin(__builtin_amdgcn_s_wait_asynccnt)
    __builtin_amdgcn_s_wait_asynccnt(N);
#else
    if (N == 0) asm volatile("s_wait_asynccnt 0x0" ::: "memory");
    else        asm volatile("s_wait_asynccnt 0x9" ::: "memory");
#endif
#endif
}

// ---- prep: split x into hi/lo bf16 planes -------------------------------
__global__ void prep_x(const float* __restrict__ x,
                       unsigned short* __restrict__ xh,
                       unsigned short* __restrict__ xl) {
    int i = blockIdx.x * blockDim.x + threadIdx.x;
    if (i < NN * FIN) {
        float v = x[i];
        unsigned short h = f2bf(v);
        xh[i] = h;
        xl[i] = f2bf(v - bf2f(h));
    }
}

// ---- prep: transpose W[h][k][d] -> Wt[h*64+d][k], hi/lo planes ----------
__global__ void prep_w(const float* __restrict__ W,
                       unsigned short* __restrict__ wh,
                       unsigned short* __restrict__ wl) {
    int i = blockIdx.x * blockDim.x + threadIdx.x;
    if (i < NH * FIN * DH) {
        int d = i % DH;
        int k = (i / DH) % FIN;
        int h = i / (DH * FIN);
        float v = W[i];
        int o = (h * DH + d) * FIN + k;
        unsigned short hi = f2bf(v);
        wh[o] = hi;
        wl[o] = f2bf(v - bf2f(hi));
    }
}

// ---- prep: zero the padded Bt planes ------------------------------------
__global__ void fill_bt(unsigned int* __restrict__ bh, unsigned int* __restrict__ bl) {
    int i = blockIdx.x * blockDim.x + threadIdx.x;
    if (i < CPAD * NN / 2) { bh[i] = 0u; bl[i] = 0u; }
}

// ---- projection: mx = x@W (hi/lo WMMA), then c2, e2, build Bt -----------
// Block = 256 thr = 8 waves = 2 row-strips x 4 heads. Each wave owns a
// 16-row x 64-col (one head) strip of mx.
__global__ void __launch_bounds__(256) proj_kernel(
    const unsigned short* __restrict__ xh, const unsigned short* __restrict__ xl,
    const unsigned short* __restrict__ wh, const unsigned short* __restrict__ wl,
    const float* __restrict__ alpha,
    unsigned short* __restrict__ bth, unsigned short* __restrict__ btl)
{
    const int tid   = threadIdx.x;
    const int wave  = tid >> 5;
    const int lane  = tid & 31;
    const int h     = wave & 3;
    const int strip = wave >> 2;
    const int r0    = blockIdx.x * 32 + strip * 16;
    const int lrow  = lane & 15;
    const int khalf = (lane < 16) ? 0 : 8;     // 16-bit A-frag K sub-chunk

    Acc acc[4];
#pragma unroll
    for (int t = 0; t < 4; ++t) acc[t].v = (v8f){0,0,0,0,0,0,0,0};

    const unsigned short* xhp = xh + (size_t)(r0 + lrow) * FIN + khalf;
    const unsigned short* xlp = xl + (size_t)(r0 + lrow) * FIN + khalf;

    for (int k0 = 0; k0 < FIN; k0 += 32) {
        BFrag ah, al;
        ah.q[0] = *(const uint4*)(xhp + k0);
        ah.q[1] = *(const uint4*)(xhp + k0 + 16);
        al.q[0] = *(const uint4*)(xlp + k0);
        al.q[1] = *(const uint4*)(xlp + k0 + 16);
#pragma unroll
        for (int t = 0; t < 4; ++t) {
            const int c = h * DH + t * 16 + lrow;
            const unsigned short* wph = wh + (size_t)c * FIN + k0 + khalf;
            const unsigned short* wpl = wl + (size_t)c * FIN + k0 + khalf;
            BFrag bh_, bl_;
            bh_.q[0] = *(const uint4*)(wph);
            bh_.q[1] = *(const uint4*)(wph + 16);
            bl_.q[0] = *(const uint4*)(wpl);
            bl_.q[1] = *(const uint4*)(wpl + 16);
            // (xhi+xlo)@(whi+wlo) ~= xhi@whi + xhi@wlo + xlo@whi
            acc[t].v = __builtin_amdgcn_wmma_f32_16x16x32_bf16(false, ah.v, false, bh_.v, (short)0, acc[t].v, false, false);
            acc[t].v = __builtin_amdgcn_wmma_f32_16x16x32_bf16(false, ah.v, false, bl_.v, (short)0, acc[t].v, false, false);
            acc[t].v = __builtin_amdgcn_wmma_f32_16x16x32_bf16(false, al.v, false, bh_.v, (short)0, acc[t].v, false, false);
        }
    }

    // c2[row] = sum_d mx[row][d]*alpha[h][d]; xor-reduce within 16-lane half
    float av[4];
#pragma unroll
    for (int t = 0; t < 4; ++t) av[t] = alpha[h * DH + t * 16 + lrow];

    float e2v[8];
#pragma unroll
    for (int v = 0; v < 8; ++v) {
        float p = acc[0].f[v]*av[0] + acc[1].f[v]*av[1]
                + acc[2].f[v]*av[2] + acc[3].f[v]*av[3];
        p += __shfl_xor(p, 1, 32);
        p += __shfl_xor(p, 2, 32);
        p += __shfl_xor(p, 4, 32);
        p += __shfl_xor(p, 8, 32);
        e2v[v] = expf(p);                       // un-stabilized exp, as reference
    }

    const int roff = (lane < 16) ? 0 : 8;       // C-layout: high half holds M=v+8

    // Bt[c][j] = e2[j]*mx[j][c] as hi/lo bf16 (packed 8 rows -> one b128 store)
#pragma unroll
    for (int t = 0; t < 4; ++t) {
        unsigned int hiw[4], low[4];
#pragma unroll
        for (int v = 0; v < 8; v += 2) {
            float a0 = e2v[v]   * acc[t].f[v];
            float a1 = e2v[v+1] * acc[t].f[v+1];
            unsigned short h0 = f2bf(a0), h1 = f2bf(a1);
            hiw[v>>1] = (unsigned int)h0 | ((unsigned int)h1 << 16);
            low[v>>1] = (unsigned int)f2bf(a0 - bf2f(h0))
                      | ((unsigned int)f2bf(a1 - bf2f(h1)) << 16);
        }
        const int c = h * DH + t * 16 + lrow;
        size_t o = (size_t)c * NN + r0 + roff;
        *(uint4*)(bth + o) = make_uint4(hiw[0], hiw[1], hiw[2], hiw[3]);
        *(uint4*)(btl + o) = make_uint4(low[0], low[1], low[2], low[3]);
    }
    // denominator column: Bt[256+h][j] = e2[j]
    if (lrow == 0) {
        unsigned int hiw[4], low[4];
#pragma unroll
        for (int v = 0; v < 8; v += 2) {
            unsigned short h0 = f2bf(e2v[v]), h1 = f2bf(e2v[v+1]);
            hiw[v>>1] = (unsigned int)h0 | ((unsigned int)h1 << 16);
            low[v>>1] = (unsigned int)f2bf(e2v[v]   - bf2f(h0))
                      | ((unsigned int)f2bf(e2v[v+1] - bf2f(h1)) << 16);
        }
        size_t o = (size_t)(OUTC + h) * NN + r0 + roff;
        *(uint4*)(bth + o) = make_uint4(hiw[0], hiw[1], hiw[2], hiw[3]);
        *(uint4*)(btl + o) = make_uint4(low[0], low[1], low[2], low[3]);
    }
}

// ---- aggregation: out = (adj @ Bt^T) with per-row division ---------------
// Block = 256 thr = 8 waves; wave = one 16-row strip x all 18 column tiles.
// Bt K-slabs (32 x 288 x {hi,lo}) are async-copied into double-buffered LDS
// once per block and shared by all 8 waves.
__global__ void __launch_bounds__(256) agg_kernel(
    const float* __restrict__ adj,
    const unsigned short* __restrict__ bth, const unsigned short* __restrict__ btl,
    float* __restrict__ out)
{
    // [buffer][plane hi/lo][col * LCS + kk], LCS=40 pads away bank conflicts
    __shared__ unsigned short smem[2][2][CPAD * LCS];

    const int tid   = threadIdx.x;
    const int wave  = tid >> 5;
    const int lane  = tid & 31;
    const int r0    = blockIdx.x * 128 + wave * 16;
    const int lrow  = lane & 15;
    const int khalf = (lane < 16) ? 0 : 8;

    Acc acc[NT];
#pragma unroll
    for (int t = 0; t < NT; ++t) acc[t].v = (v8f){0,0,0,0,0,0,0,0};

    const float* ap = adj + (size_t)(r0 + lrow) * NN + khalf;

    // 2304 16B-chunks per slab (2 planes * 288 cols * 4), exactly 9/thread
    auto issue_slab = [&](int k0, int buf) {
#pragma unroll
        for (int j = 0; j < 9; ++j) {
            int ch    = tid + j * 256;
            int plane = (ch >= 1152) ? 1 : 0;
            int r     = ch - plane * 1152;
            int c     = r >> 2;
            int q     = r & 3;
            const unsigned short* src =
                (plane ? btl : bth) + (size_t)c * NN + k0 + q * 8;
            unsigned short* dst = &smem[buf][plane][c * LCS + q * 8];
            async_cp16(src, dst);
        }
    };

    const int KSTEPS = NN / 32;
    issue_slab(0, 0);

    for (int ks = 0; ks < KSTEPS; ++ks) {
        const int k0  = ks * 32;
        const int buf = ks & 1;

        // A fragment: adj rows (global, per-wave private), truncating bf16 pack
        float4 f0 = *(const float4*)(ap + k0);
        float4 f1 = *(const float4*)(ap + k0 + 4);
        float4 f2 = *(const float4*)(ap + k0 + 16);
        float4 f3 = *(const float4*)(ap + k0 + 20);
        __builtin_prefetch((const void*)(ap + k0 + 128), 0, 1);  // stream ahead
        BFrag a;
        a.u[0] = pack2_trunc(f0.x, f0.y);
        a.u[1] = pack2_trunc(f0.z, f0.w);
        a.u[2] = pack2_trunc(f1.x, f1.y);
        a.u[3] = pack2_trunc(f1.z, f1.w);
        a.u[4] = pack2_trunc(f2.x, f2.y);
        a.u[5] = pack2_trunc(f2.z, f2.w);
        a.u[6] = pack2_trunc(f3.x, f3.y);
        a.u[7] = pack2_trunc(f3.z, f3.w);

        if (ks + 1 < KSTEPS) {           // prefetch next slab, then wait for
            issue_slab(k0 + 32, buf ^ 1);//   current (<=9 outstanding = next's)
            async_wait<9>();
        } else {
            async_wait<0>();
        }
        __syncthreads();                 // slab `buf` visible to all waves

#pragma unroll
        for (int t = 0; t < NT; ++t) {
            const int c = t * 16 + lrow;
            BFrag bh_, bl_;
            bh_.q[0] = *(const uint4*)&smem[buf][0][c * LCS + khalf];
            bh_.q[1] = *(const uint4*)&smem[buf][0][c * LCS + khalf + 16];
            bl_.q[0] = *(const uint4*)&smem[buf][1][c * LCS + khalf];
            bl_.q[1] = *(const uint4*)&smem[buf][1][c * LCS + khalf + 16];
            acc[t].v = __builtin_amdgcn_wmma_f32_16x16x32_bf16(false, a.v, false, bh_.v, (short)0, acc[t].v, false, false);
            acc[t].v = __builtin_amdgcn_wmma_f32_16x16x32_bf16(false, a.v, false, bl_.v, (short)0, acc[t].v, false, false);
        }
        __syncthreads();                 // all waves done before buf is re-filled
    }

    const int roff = (lane < 16) ? 0 : 8;
    // tile 16 holds denominator cols 256..259 (lane lrow==h); shuffle + divide
#pragma unroll
    for (int t = 0; t < 16; ++t) {
        const int head = t >> 2;
        const int src  = (lane & 16) + head;   // same half, lrow == head
        const int c    = t * 16 + lrow;
#pragma unroll
        for (int v = 0; v < 8; ++v) {
            float den = __shfl(acc[16].f[v], src, 32);
            out[(size_t)(r0 + roff + v) * OUTC + c] = acc[t].f[v] / den;
        }
    }
}

extern "C" void kernel_launch(void* const* d_in, const int* in_sizes, int n_in,
                              void* d_out, int out_size, void* d_ws, size_t ws_size,
                              hipStream_t stream) {
    const float* x     = (const float*)d_in[0];
    const float* adj   = (const float*)d_in[1];
    const float* W     = (const float*)d_in[2];
    // d_in[3] = alpha_origin: cancels in softmax normalization, unused.
    const float* alpha = (const float*)d_in[4];
    float* out = (float*)d_out;

    unsigned short* xh  = (unsigned short*)d_ws;
    unsigned short* xl  = xh  + (size_t)NN * FIN;
    unsigned short* wh  = xl  + (size_t)NN * FIN;
    unsigned short* wl  = wh  + (size_t)NH * FIN * DH;
    unsigned short* bth = wl  + (size_t)NH * FIN * DH;
    unsigned short* btl = bth + (size_t)CPAD * NN;
    // total ws use: ~13 MB

    prep_x <<<(NN * FIN + 255) / 256, 256, 0, stream>>>(x, xh, xl);
    prep_w <<<(NH * FIN * DH + 255) / 256, 256, 0, stream>>>(W, wh, wl);
    fill_bt<<<(CPAD * NN / 2 + 255) / 256, 256, 0, stream>>>((unsigned int*)bth, (unsigned int*)btl);
    proj_kernel<<<NN / 32, 256, 0, stream>>>(xh, xl, wh, wl, alpha, bth, btl);
    agg_kernel <<<NN / 128, 256, 0, stream>>>(adj, bth, btl, out);
}
